// Att_H_47571057771075
// MI455X (gfx1250) — matrix operand order repaired and checked by reading, not compiled, not run on hardware
//
#include <hip/hip_runtime.h>
#include <hip/hip_bf16.h>

// ---------------------------------------------------------------------------
// MI455X (gfx1250, wave32). bf16 WMMA (v_wmma_f32_16x16x32_bf16) everywhere,
// f32 accumulate. All operands pre-converted to bf16 once; all fragment loads
// are contiguous 16B (global_load_b128 / ds_load_b128). x (134MB f32) streamed
// once through double-buffered LDS tiles with reg-level prefetch pipelining.
// ---------------------------------------------------------------------------

typedef __attribute__((ext_vector_type(16))) __bf16 v16bf;
typedef __attribute__((ext_vector_type(8)))  __bf16 v8bf;
typedef __attribute__((ext_vector_type(8)))  float  v8f;

#define NA 256   // sequence length n
#define RR 64    // rank r

static __device__ __forceinline__ __bf16 tobf(float f) {
    // round-to-nearest-even f32 -> bf16
    unsigned u = __float_as_uint(f);
    u += 0x7FFFu + ((u >> 16) & 1u);
    unsigned short h = (unsigned short)(u >> 16);
    __bf16 r;
    __builtin_memcpy(&r, &h, 2);
    return r;
}

static __device__ __forceinline__ float sigm(float v) {
    return 1.0f / (1.0f + __expf(-v));
}

static __device__ __forceinline__ v16bf cat16(v8bf lo, v8bf hi) {
    return __builtin_shufflevector(lo, hi, 0, 1, 2, 3, 4, 5, 6, 7,
                                   8, 9, 10, 11, 12, 13, 14, 15);
}

// ---------------------------------------------------------------------------
// One-time f32 -> bf16 conversion (weights).
// ---------------------------------------------------------------------------
__global__ void cvt_kernel(const float* __restrict__ src, __bf16* __restrict__ dst, int n) {
    int i = blockIdx.x * 256 + threadIdx.x;
    if (i < n) dst[i] = tobf(src[i]);
}

// ---------------------------------------------------------------------------
// scale[b] = sqrt(1 + sum_n (L[b,0,n] >= 1))
// ---------------------------------------------------------------------------
__global__ void scale_kernel(const float* __restrict__ L, float* __restrict__ scale) {
    __shared__ float red[256];
    const int b = blockIdx.x;
    const int t = threadIdx.x;
    red[t] = (L[(long)b * 2 * NA + t] >= 1.0f) ? 1.0f : 0.0f;
    __syncthreads();
    for (int s = 128; s >= 1; s >>= 1) {
        if (t < s) red[t] += red[t + s];
        __syncthreads();
    }
    if (t == 0) scale[b] = sqrtf(red[0] + 1.0f);
}

// ---------------------------------------------------------------------------
// Fused QKV projection: sigmoid(concat(Wq,Wk,Wv)[192xK] @ X[KxNA])
// 12 waves share one 16-col x tile (bf16, column-major in LDS, ping-pong
// buffers, reg prefetch pipelining). Weights already bf16 row-major.
// Outputs: Q,K stored (NA x RR) n-major (v8bf stores); V stored (RR x NA).
// ---------------------------------------------------------------------------
__global__ void qkv_proj_kernel(const __bf16* __restrict__ Wq,
                                const __bf16* __restrict__ Wk,
                                const __bf16* __restrict__ Wv,
                                const float* __restrict__ X, long xBatch, int K,
                                __bf16* __restrict__ Qn, __bf16* __restrict__ Kn,
                                __bf16* __restrict__ Vr, long pBatch) {
    __shared__ __bf16 Xs[2][16 * 32];   // [col][k], ping-pong
    const int b    = blockIdx.y;
    const int col0 = blockIdx.x * 16;
    const int tid  = threadIdx.x;
    const int wave = tid >> 5;
    const int lane = tid & 31;
    const float* Xb = X + (long)b * xBatch;

    // A-fragment: 16-bit A 16x32 layout (M = lane&15, K halves by lane>>4)
    const int mrow = wave * 16 + (lane & 15);        // 0..191
    const int wsel = mrow >> 6;
    const int lr   = mrow & 63;
    const __bf16* W = (wsel == 0) ? Wq : (wsel == 1 ? Wk : Wv);
    const __bf16* arow = W + (long)lr * K;
    const int kh   = (lane >> 4) * 8;
    const int kb   = (lane >> 4) * 16;
    const int bcol = lane & 15;

    // staging: waves 0..7 (256 threads), one float2 each per tile
    const bool stager = (tid < 256);
    const int  sk = tid >> 3;          // k row 0..31
    const int  sc = (tid & 7) << 1;    // col pair 0,2,...,14

    float2 pf;
    if (stager) {
        pf = *(const float2*)(Xb + (long)sk * NA + col0 + sc);
        Xs[0][(sc    ) * 32 + sk] = tobf(pf.x);
        Xs[0][(sc + 1) * 32 + sk] = tobf(pf.y);
    }
    __syncthreads();

    v8f acc = {};
    int cur = 0;
    for (int k0 = 0; k0 < K; k0 += 32) {
        const bool more = (k0 + 32 < K);
        if (more && stager)   // issue next-tile global loads before the WMMA
            pf = *(const float2*)(Xb + (long)(k0 + 32 + sk) * NA + col0 + sc);

        v8bf alo = *(const v8bf*)(arow + k0 + kh);
        v8bf ahi = *(const v8bf*)(arow + k0 + kh + 16);
        const __bf16* bp = &Xs[cur][bcol * 32 + kh];
        v8bf blo = *(const v8bf*)bp;
        v8bf bhi = *(const v8bf*)(bp + 16);
        v16bf areg = cat16(alo, ahi);
        v16bf breg = cat16(blo, bhi);
        acc = __builtin_amdgcn_wmma_f32_16x16x32_bf16(false, areg, false, breg,
                                                      (short)0, acc, false, false);

        if (more && stager) {   // drain prefetch into the other LDS buffer
            Xs[cur ^ 1][(sc    ) * 32 + sk] = tobf(pf.x);
            Xs[cur ^ 1][(sc + 1) * 32 + sk] = tobf(pf.y);
        }
        __syncthreads();
        cur ^= 1;
    }

    // D layout: col = lane&15, row_local = v + 8*(lane>>4). The 8 rows per
    // lane are consecutive -> vectorized v8bf store for Q/K (n-major).
    const int n   = col0 + (lane & 15);
    const int mg0 = wave * 16 + 8 * (lane >> 4);
    const int ws2 = mg0 >> 6;
    const int lr0 = mg0 & 63;
    if (ws2 <= 1) {
        v8bf ov;
        #pragma unroll
        for (int v = 0; v < 8; ++v) ov[v] = tobf(sigm(acc[v]));
        __bf16* dst = (ws2 == 0 ? Qn : Kn) + (long)b * pBatch + (long)n * RR + lr0;
        *(v8bf*)dst = ov;
    } else {
        #pragma unroll
        for (int v = 0; v < 8; ++v)
            Vr[(long)b * pBatch + (long)(lr0 + v) * NA + n] = tobf(sigm(acc[v]));
    }
}

// ---------------------------------------------------------------------------
// Generic bf16 WMMA GEMM, always B-transposed storage:
//   C[MxN] = act( alpha * A[MxK] @ Bt^T ),  Bt stored (N x K) row-major.
// Every lane's A/B fragment = two contiguous 16B global loads.
// Output: f32 (Cf) or bf16 (Cb). act: 0 none, 2 silu. alpha = 1/scaleArr[b].
// Block = 4 waves along N, each owning one 16x16 tile.
// ---------------------------------------------------------------------------
__global__ void wmma_gemm_bt_kernel(const __bf16* __restrict__ A, long aBatch,
                                    const __bf16* __restrict__ Bt, long bBatch,
                                    float* __restrict__ Cf, __bf16* __restrict__ Cb,
                                    long cBatch, int M, int N, int K, int act,
                                    const float* __restrict__ scaleArr) {
    const int b    = blockIdx.z;
    const int lane = threadIdx.x & 31;
    const int wave = threadIdx.x >> 5;
    const int col0 = (blockIdx.x * 4 + wave) * 16;
    const int row0 = blockIdx.y * 16;
    if (col0 >= N) return;  // wave-uniform

    const __bf16* arow = A  + (long)b * aBatch + (long)(row0 + (lane & 15)) * K;
    const __bf16* brow = Bt + (long)b * bBatch + (long)(col0 + (lane & 15)) * K;
    const int kh = (lane >> 4) * 8;
    const int kb = (lane >> 4) * 16;

    v8f acc = {};
    #pragma unroll 2
    for (int k0 = 0; k0 < K; k0 += 32) {
        v8bf alo = *(const v8bf*)(arow + k0 + kh);
        v8bf ahi = *(const v8bf*)(arow + k0 + kh + 16);
        v8bf blo = *(const v8bf*)(brow + k0 + kh);
        v8bf bhi = *(const v8bf*)(brow + k0 + kh + 16);
        v16bf areg = cat16(alo, ahi);
        v16bf breg = cat16(blo, bhi);
        acc = __builtin_amdgcn_wmma_f32_16x16x32_bf16(false, areg, false, breg,
                                                      (short)0, acc, false, false);
    }

    const float alpha = scaleArr ? (1.0f / scaleArr[b]) : 1.0f;
    const int bcol = col0 + (lane & 15);
    #pragma unroll
    for (int v = 0; v < 8; ++v) {
        int   mg  = row0 + v + 8 * (lane >> 4);
        float val = acc[v] * alpha;
        if (act == 2) val = val * sigm(val);   // silu
        long  idx = (long)b * cBatch + (long)mg * N + bcol;
        if (Cb) Cb[idx] = tobf(val);
        else    Cf[idx] = val;
    }
}

// ---------------------------------------------------------------------------
// Row softmax over NA=256 (one wave per row); f32 in, bf16 out.
// ---------------------------------------------------------------------------
__global__ void softmax_kernel(const float* __restrict__ Sf, __bf16* __restrict__ Sb,
                               long sBatch) {
    const int b    = blockIdx.y;
    const int row  = blockIdx.x * 8 + (threadIdx.x >> 5);
    const int lane = threadIdx.x & 31;
    const float* p = Sf + (long)b * sBatch + (long)row * NA;
    __bf16*      q = Sb + (long)b * sBatch + (long)row * NA;
    float v[8];
    float mx = -1e30f;
    #pragma unroll
    for (int i = 0; i < 8; ++i) { v[i] = p[lane + 32 * i]; mx = fmaxf(mx, v[i]); }
    #pragma unroll
    for (int s = 16; s >= 1; s >>= 1) mx = fmaxf(mx, __shfl_xor(mx, s, 32));
    float sum = 0.0f;
    #pragma unroll
    for (int i = 0; i < 8; ++i) { v[i] = __expf(v[i] - mx); sum += v[i]; }
    #pragma unroll
    for (int s = 16; s >= 1; s >>= 1) sum += __shfl_xor(sum, s, 32);
    const float inv = 1.0f / sum;
    #pragma unroll
    for (int i = 0; i < 8; ++i) q[lane + 32 * i] = tobf(v[i] * inv);
}

// ---------------------------------------------------------------------------
// Final quadratic form (see round-0 derivation). M is f32 (64 x 256).
// ---------------------------------------------------------------------------
__global__ void quad_kernel(const float* __restrict__ Mf, long mBatch,
                            const float* __restrict__ x, long xBatch,
                            float* __restrict__ out) {
    __shared__ float sD[4][NA];
    __shared__ float red[256];
    const int b = blockIdx.x;
    const int t = threadIdx.x;
    const float* Mb = Mf + (long)b * mBatch;

    float d[4] = {0, 0, 0, 0}, dpp = 0.0f;
    #pragma unroll
    for (int g = 0; g < 4; ++g)
        for (int r = 0; r < 8; ++r) { float m = Mb[(g * 8 + r) * NA + t]; d[g] += m * m; }
    for (int r = 32; r < 64; ++r) { float m = Mb[r * NA + t]; dpp += m * m; }
    sD[0][t] = d[0]; sD[1][t] = d[1]; sD[2][t] = d[2]; sD[3][t] = d[3];
    __syncthreads();

    const int n     = t & 127;
    const int cbase = (t >> 7) * 2;
    const float* xb = x + (long)b * xBatch;
    float part = dpp;
    #pragma unroll
    for (int cc = 0; cc < 2; ++cc) {
        int c = cbase + cc;
        float q1 = xb[(long)c * NA + n];
        float q2 = xb[(long)c * NA + n + 128];
        int idx = 2 * n + (c >> 1);
        part += sD[0][idx] * q1 * q1 + sD[1][idx] * q1 * q2 +
                sD[2][idx] * q2 * q1 + sD[3][idx] * q2 * q2;
    }
    red[t] = part;
    __syncthreads();
    for (int s = 128; s >= 1; s >>= 1) {
        if (t < s) red[t] += red[t + s];
        __syncthreads();
    }
    if (t == 0) out[b] = red[0];
}

// ---------------------------------------------------------------------------
extern "C" void kernel_launch(void* const* d_in, const int* in_sizes, int n_in,
                              void* d_out, int out_size, void* d_ws, size_t ws_size,
                              hipStream_t stream) {
    const float* x = (const float*)d_in[0];
    const float* L = (const float*)d_in[1];
    const float* Wqf[3] = {(const float*)d_in[2], (const float*)d_in[5], (const float*)d_in[8]};
    const float* Wkf[3] = {(const float*)d_in[3], (const float*)d_in[6], (const float*)d_in[9]};
    const float* Wvf[3] = {(const float*)d_in[4], (const float*)d_in[7], (const float*)d_in[10]};
    const float* Wof[3] = {(const float*)d_in[11], (const float*)d_in[12], (const float*)d_in[13]};
    float* out = (float*)d_out;

    const int  B  = 64;
    const long P  = (long)RR * NA;   // 16384
    const long SS = (long)NA * NA;   // 65536

    // ---- workspace carve-up (f32 region then 16B-aligned bf16 region) ----
    float* wsf   = (float*)d_ws;
    float* scale = wsf;                       // 64
    float* Sf    = wsf + 64;                  // B*SS f32 scores
    float* Mb    = Sf + B * SS;               // B*P  f32 layer output
    __bf16* Qbf  = (__bf16*)(Mb + B * P);     // B*P  (NA x RR)
    __bf16* Kbf  = Qbf + B * P;               // B*P  (NA x RR)
    __bf16* Vbf  = Kbf + B * P;               // B*P  (RR x NA)
    __bf16* Sbf  = Vbf + B * P;               // B*SS (NA x NA)
    __bf16* Obf  = Sbf + B * SS;              // B*P  (NA x RR)
    __bf16* Wq0  = Obf + B * P;               // 64*2048
    __bf16* Wk0  = Wq0 + 64 * 2048;
    __bf16* Wv0  = Wk0 + 64 * 2048;
    __bf16* Wsm  = Wv0 + 64 * 2048;           // 9 x 4096 small weights
    __bf16* WqL[3] = {Wq0, Wsm + 0 * 4096, Wsm + 3 * 4096};
    __bf16* WkL[3] = {Wk0, Wsm + 1 * 4096, Wsm + 4 * 4096};
    __bf16* WvL[3] = {Wv0, Wsm + 2 * 4096, Wsm + 5 * 4096};
    __bf16* WoL[3] = {Wsm + 6 * 4096, Wsm + 7 * 4096, Wsm + 8 * 4096};

    // ---- one-time bf16 weight conversion ----
    cvt_kernel<<<dim3(512), dim3(256), 0, stream>>>(Wqf[0], WqL[0], 64 * 2048);
    cvt_kernel<<<dim3(512), dim3(256), 0, stream>>>(Wkf[0], WkL[0], 64 * 2048);
    cvt_kernel<<<dim3(512), dim3(256), 0, stream>>>(Wvf[0], WvL[0], 64 * 2048);
    for (int l = 1; l < 3; ++l) {
        cvt_kernel<<<dim3(16), dim3(256), 0, stream>>>(Wqf[l], WqL[l], 4096);
        cvt_kernel<<<dim3(16), dim3(256), 0, stream>>>(Wkf[l], WkL[l], 4096);
        cvt_kernel<<<dim3(16), dim3(256), 0, stream>>>(Wvf[l], WvL[l], 4096);
    }
    for (int l = 0; l < 3; ++l)
        cvt_kernel<<<dim3(16), dim3(256), 0, stream>>>(Wof[l], WoL[l], 4096);

    scale_kernel<<<dim3(B), dim3(256), 0, stream>>>(L, scale);

    for (int l = 0; l < 3; ++l) {
        const float* Xl = (l == 0) ? x : Mb;
        const long  xb  = (l == 0) ? 2048L * NA : P;
        const int   K   = (l == 0) ? 2048 : 64;

        // Q,K,V = sigmoid(W @ X); layer 1 streams the 134MB x exactly once
        qkv_proj_kernel<<<dim3(16, B), dim3(384), 0, stream>>>(
            WqL[l], WkL[l], WvL[l], Xl, xb, K, Qbf, Kbf, Vbf, P);

        // S = (Q @ K^T)/scale[b] : A=Qbf(256x64), Bt=Kbf(256x64) -> f32
        wmma_gemm_bt_kernel<<<dim3(4, 16, B), dim3(128), 0, stream>>>(
            Qbf, P, Kbf, P, Sf, nullptr, SS, NA, NA, RR, 0, scale);

        softmax_kernel<<<dim3(32, B), dim3(256), 0, stream>>>(Sf, Sbf, SS);

        // O = A @ V : A=Sbf(256x256), Bt=Vbf(64x256) -> bf16 (NA x RR)
        wmma_gemm_bt_kernel<<<dim3(1, 16, B), dim3(128), 0, stream>>>(
            Sbf, SS, Vbf, P, nullptr, Obf, P, NA, RR, NA, 0, nullptr);

        // M = silu(Wo @ O^T) : A=Wo(64x64), Bt=Obf(256x64) -> f32 (RR x NA)
        wmma_gemm_bt_kernel<<<dim3(4, 4, B), dim3(128), 0, stream>>>(
            WoL[l], 0, Obf, P, Mb, nullptr, P, RR, NA, RR, 2, nullptr);
    }

    quad_kernel<<<dim3(B), dim3(256), 0, stream>>>(Mb, P, x, 2048L * NA, out);
}
